// Conv2dOrion_64226940944754
// MI455X (gfx1250) — matrix operand order repaired and checked by reading, not compiled, not run on hardware
//
#include <hip/hip_runtime.h>

typedef float    v4f  __attribute__((ext_vector_type(4)));
typedef float    v8f  __attribute__((ext_vector_type(8)));
typedef _Float16 v2h  __attribute__((ext_vector_type(2)));
typedef _Float16 v4h  __attribute__((ext_vector_type(4)));
typedef _Float16 v8h  __attribute__((ext_vector_type(8)));
typedef _Float16 v16h __attribute__((ext_vector_type(16)));

#define KC 64     // K-chunk: 2 wmma k-steps of 32
#define LS 72     // LDS K-stride (f16): 144B rows -> conflict-free b128 frag loads
#define MT 128    // Cout tile per workgroup
#define NT 64     // output positions per workgroup
#define WELEMS (256 * 2304)           // weight element count
#define WS_NEED ((size_t)WELEMS * 4)  // bytes for wh+wl in d_ws

// A-fragment (16x32 f16): lane<16 holds K = k0+{0..7} and k0+16+{0..7}; lane>=16 at +8.
__device__ __forceinline__ v16h ld_frag_a(const _Float16* row, int k0, int hi) {
  const int o = k0 + 8 * hi;
  v8h lo = *(const v8h*)(row + o);
  v8h hh = *(const v8h*)(row + o + 16);
  return __builtin_shufflevector(lo, hh, 0,1,2,3,4,5,6,7,8,9,10,11,12,13,14,15);
}
// B-fragment (32x16 f16): lane<16 holds K = k0+{0..7} and k0+16+{0..7}; lane>=16 at +8.
__device__ __forceinline__ v16h ld_frag_b(const _Float16* row, int k0, int hi) {
  const int o = k0 + 8 * hi;
  v8h lo = *(const v8h*)(row + o);
  v8h hh = *(const v8h*)(row + o + 16);
  return __builtin_shufflevector(lo, hh, 0,1,2,3,4,5,6,7,8,9,10,11,12,13,14,15);
}

#define WMMA(A, B, C) \
  __builtin_amdgcn_wmma_f32_16x16x32_f16(false, (A), false, (B), (short)0, (C), false, false)

// ---- pre-kernel: split fp32 weights into f16 hi/lo planes in workspace ----
__global__ __launch_bounds__(256)
void split_weights(const float* __restrict__ w, _Float16* __restrict__ wh,
                   _Float16* __restrict__ wl) {
  const int i = (blockIdx.x * 256 + threadIdx.x) * 4;  // WELEMS/4 threads
  v4f v = *(const v4f*)(w + i);
  v4h vh, vl;
  #pragma unroll
  for (int j = 0; j < 4; ++j) {
    const _Float16 h = (_Float16)v[j];
    vh[j] = h;
    vl[j] = (_Float16)(v[j] - (float)h);
  }
  *(v4h*)(wh + i) = vh;
  *(v4h*)(wl + i) = vl;
}

template <bool PRESPLIT>
__global__ __launch_bounds__(256)
void conv2d_s2_wmma_f16x3(const float* __restrict__ x,      // [16,256,64,64]
                          const float* __restrict__ w,      // [256][2304] fp32
                          const _Float16* __restrict__ wh,  // [256][2304] f16 hi (ws)
                          const _Float16* __restrict__ wl,  // [256][2304] f16 lo (ws)
                          const float* __restrict__ bias,
                          float* __restrict__ out) {        // [16,256,1024] permuted
  __shared__ _Float16 Ah[MT * LS];
  __shared__ _Float16 Al[MT * LS];
  __shared__ _Float16 Bh[NT * LS];
  __shared__ _Float16 Bl[NT * LS];

  int bid = blockIdx.x;
  const int mTile  = bid & 1;  bid >>= 1;
  const int nTileV = bid & 3;  bid >>= 2;
  const int blk    = bid & 3;  bid >>= 2;
  const int nb     = bid;
  const int si = blk >> 1, sj = blk & 1;
  const int coBase = mTile * MT;

  const int tid  = threadIdx.x;
  const int lane = tid & 31;
  const int lm   = lane & 15;
  const int hi   = lane >> 4;
  const int wave = tid >> 5;
  const int mOff = (wave & 3) * 32;
  const int nOff = (wave >> 2) * 32;

  // B staging: thread -> (k-pair, 8 positions sharing one (ci,kh,kw) decode)
  const int kkA  = (tid & 31) * 2;        // k-pair base within chunk
  const int pg   = tid >> 5;              // 0..7 position group
  const int iiB  = pg >> 1;               // parity-grid row within tile
  const int jjB  = (pg & 1) * 8;          // first column of the 8-position run
  const int rbaseB = 4 * (nTileV * 4 + iiB) + 2 * si - 1;  // in [-1,61]
  const int cbase0 = 4 * jjB + 2 * sj - 1;                 // c = cbase0 + 4v + kw
  const float* xn = x + ((size_t)nb << 20);

  const _Float16* ah0 = &Ah[(mOff +      lm) * LS];
  const _Float16* ah1 = &Ah[(mOff + 16 + lm) * LS];
  const _Float16* al0 = &Al[(mOff +      lm) * LS];
  const _Float16* al1 = &Al[(mOff + 16 + lm) * LS];
  const _Float16* bh0 = &Bh[(nOff +      lm) * LS];
  const _Float16* bh1 = &Bh[(nOff + 16 + lm) * LS];
  const _Float16* bl0 = &Bl[(nOff +      lm) * LS];
  const _Float16* bl1 = &Bl[(nOff + 16 + lm) * LS];

  v8f acc00 = {}, acc01 = {}, acc10 = {}, acc11 = {};

  for (int kc = 0; kc < 2304 / KC; ++kc) {
    __syncthreads();
    // ---- stage A ----
    if (PRESPLIT) {
      // pure f16 b128 copies: global (hi/lo planes) -> LDS
      const size_t gbase = (size_t)coBase * 2304 + kc * KC;
      #pragma unroll
      for (int idx = tid; idx < MT * (KC / 8); idx += 256) {   // 4 iters
        const int row = idx >> 3, c8 = idx & 7;
        const size_t g = gbase + (size_t)row * 2304 + c8 * 8;
        *(v8h*)&Ah[row * LS + c8 * 8] = *(const v8h*)(wh + g);
        *(v8h*)&Al[row * LS + c8 * 8] = *(const v8h*)(wl + g);
      }
    } else {
      // fallback: load fp32 weights and split in-kernel
      const float* wbase = w + (size_t)coBase * 2304 + kc * KC;
      #pragma unroll
      for (int idx = tid; idx < MT * (KC / 4); idx += 256) {
        const int row = idx >> 4, c4 = idx & 15;
        v4f v = *(const v4f*)(wbase + (size_t)row * 2304 + c4 * 4);
        v4h vh, vl;
        #pragma unroll
        for (int j = 0; j < 4; ++j) {
          const _Float16 h = (_Float16)v[j];
          vh[j] = h;
          vl[j] = (_Float16)(v[j] - (float)h);
        }
        *(v4h*)&Ah[row * LS + c4 * 4] = vh;
        *(v4h*)&Al[row * LS + c4 * 4] = vl;
      }
    }
    // ---- stage B: decode 2 k's once, reuse across 8 positions ----
    {
      int   off[2], kwv[2];
      bool  rOK[2];
      #pragma unroll
      for (int t = 0; t < 2; ++t) {
        const int k   = kc * KC + kkA + t;
        const int ci  = (unsigned)k / 9u;
        const int tap = k - ci * 9;
        const int kh  = (unsigned)tap / 3u;
        const int kw  = tap - kh * 3;
        const int r   = rbaseB + kh;           // <= 63 by construction
        rOK[t] = (r >= 0);
        kwv[t] = kw;
        off[t] = (ci << 12) + r * 64 + kw;     // + (cbase0 + 4v) below
      }
      #pragma unroll
      for (int v = 0; v < 8; ++v) {
        const int cb = cbase0 + 4 * v;
        float f0 = (rOK[0] && (cb + kwv[0] >= 0)) ? xn[off[0] + cb] : 0.0f;
        float f1 = (rOK[1] && (cb + kwv[1] >= 0)) ? xn[off[1] + cb] : 0.0f;
        const _Float16 h0 = (_Float16)f0, h1 = (_Float16)f1;
        v2h vh = {h0, h1};
        v2h vl = {(_Float16)(f0 - (float)h0), (_Float16)(f1 - (float)h1)};
        const int addr = (pg * 8 + v) * LS + kkA;
        *(v2h*)&Bh[addr] = vh;
        *(v2h*)&Bl[addr] = vl;
      }
    }
    __syncthreads();
    // ---- compute: 2 k-steps of 32; f16x3 split => 12 XDL WMMAs per step ----
    #pragma unroll
    for (int ks = 0; ks < 2; ++ks) {
      const int k0 = ks * 32;
      v16h a0h = ld_frag_a(ah0, k0, hi);
      v16h a1h = ld_frag_a(ah1, k0, hi);
      v16h a0l = ld_frag_a(al0, k0, hi);
      v16h a1l = ld_frag_a(al1, k0, hi);
      v16h b0h = ld_frag_b(bh0, k0, hi);
      v16h b1h = ld_frag_b(bh1, k0, hi);
      v16h b0l = ld_frag_b(bl0, k0, hi);
      v16h b1l = ld_frag_b(bl1, k0, hi);
      acc00 = WMMA(a0h, b0h, acc00);
      acc01 = WMMA(a0h, b1h, acc01);
      acc10 = WMMA(a1h, b0h, acc10);
      acc11 = WMMA(a1h, b1h, acc11);
      acc00 = WMMA(a0h, b0l, acc00);
      acc01 = WMMA(a0h, b1l, acc01);
      acc10 = WMMA(a1h, b0l, acc10);
      acc11 = WMMA(a1h, b1l, acc11);
      acc00 = WMMA(a0l, b0h, acc00);
      acc01 = WMMA(a0l, b1h, acc01);
      acc10 = WMMA(a1l, b0h, acc10);
      acc11 = WMMA(a1l, b1h, acc11);
    }
  }

  // ---- epilogue: bias add, store directly in permuted layout ----
  const int qBase = blk * 256 + nTileV * 64 + nOff;
  #pragma unroll
  for (int mSub = 0; mSub < 2; ++mSub) {
    #pragma unroll
    for (int nSub = 0; nSub < 2; ++nSub) {
      v8f acc = (mSub == 0) ? (nSub == 0 ? acc00 : acc01)
                            : (nSub == 0 ? acc10 : acc11);
      const int q = qBase + nSub * 16 + lm;
      #pragma unroll
      for (int r = 0; r < 8; ++r) {
        const int co = coBase + mOff + mSub * 16 + r + 8 * hi;
        out[((size_t)(nb * 256 + co) << 10) + q] = acc[r] + bias[co];
      }
    }
  }
}

extern "C" void kernel_launch(void* const* d_in, const int* in_sizes, int n_in,
                              void* d_out, int out_size, void* d_ws, size_t ws_size,
                              hipStream_t stream) {
  const float* x    = (const float*)d_in[0];
  const float* w    = (const float*)d_in[1];
  const float* bias = (const float*)d_in[2];
  float* out = (float*)d_out;
  (void)in_sizes; (void)n_in; (void)out_size;

  dim3 grid(16 * 4 * 4 * 2);  // batches * parity blocks * N-tiles * M-tiles
  dim3 block(256);

  if (ws_size >= WS_NEED && d_ws != nullptr) {
    _Float16* wh = (_Float16*)d_ws;
    _Float16* wl = wh + WELEMS;
    split_weights<<<dim3(WELEMS / 4 / 256), block, 0, stream>>>(w, wh, wl);
    conv2d_s2_wmma_f16x3<true><<<grid, block, 0, stream>>>(x, w, wh, wl, bias, out);
  } else {
    conv2d_s2_wmma_f16x3<false><<<grid, block, 0, stream>>>(x, w, nullptr, nullptr, bias, out);
  }
}
